// BiLSTMCRF_39840116637935
// MI455X (gfx1250) — compile-verified, matrix operand order checked by reading, not run on hardware
//
#include <hip/hip_runtime.h>
#include <hip/hip_bf16.h>
#include <math.h>

typedef __attribute__((ext_vector_type(16))) _Float16 v16h;
typedef __attribute__((ext_vector_type(8)))  float    v8f;

#define WMMA_F16(a, b, c) \
  __builtin_amdgcn_wmma_f32_16x16x32_f16(false, (a), false, (b), (short)0, (c), false, false)

// ---- problem dims ----
constexpr int B_    = 32;
constexpr int T_    = 256;
constexpr int BT    = 8192;   // B*T
constexpr int C_    = 16;     // chars per token
constexpr int E_    = 300;    // word emb
constexpr int Ce_   = 64;     // char emb
constexpr int Hc_   = 64;     // char hidden
constexpr int Hw_   = 256;    // word hidden
constexpr int L_    = 32;     // labels
constexpr int Ein_  = 428;    // E + 2*Hc
constexpr int KPAD  = 448;    // Ein padded to multiple of 32
constexpr int GW_   = 1024;   // 4*Hw
// ----------------------

// ---- fast scalar math on CDNA5 hardware units -----------------------------
__device__ inline float fast_exp2(float x) {
#if __has_builtin(__builtin_amdgcn_exp2f)
  return __builtin_amdgcn_exp2f(x);   // v_exp_f32
#else
  return exp2f(x);
#endif
}
__device__ inline float fast_rcp(float x) {
#if __has_builtin(__builtin_amdgcn_rcpf)
  return __builtin_amdgcn_rcpf(x);    // v_rcp_f32
#else
  return 1.0f / x;
#endif
}
__device__ inline float sigmoidf(float x) {
  // 1/(1+2^(-x*log2e))
  return fast_rcp(1.0f + fast_exp2(-1.4426950408889634f * x));
}
__device__ inline float tanh_fast(float x) {
#if __has_builtin(__builtin_amdgcn_tanhf)
  return __builtin_amdgcn_tanhf(x);   // v_tanh_f32 (gfx1250 TRANS op)
#else
  return 2.0f * fast_rcp(1.0f + fast_exp2(-2.8853900817779268f * x)) - 1.0f;
#endif
}
// ---------------------------------------------------------------------------

__device__ inline v8f zero8() { v8f z = {0.f,0.f,0.f,0.f,0.f,0.f,0.f,0.f}; return z; }

// A/B fragment for V_WMMA_F32_16X16X32_F16 from a row-major f16 array.
// A is 16x32 (MxK) row-major with leading dim ld.
// B (32x16, KxN) uses the same loader on data stored as [n][k] (ld = K stride).
// ISA 7.12.2: lanes 0-15 -> row, K = {0..7,16..23}; lanes 16-31 -> K = {8..15,24..31}.
__device__ inline v16h frag_h(const _Float16* base, int ld) {
  int lane = threadIdx.x & 31;
  int r    = lane & 15;
  int kb   = (lane & 16) ? 8 : 0;
  const _Float16* p = base + (long)r * ld + kb;
  v16h v;
#pragma unroll
  for (int i = 0; i < 8; ++i) v[i] = p[i];
#pragma unroll
  for (int i = 0; i < 8; ++i) v[8 + i] = p[16 + i];
  return v;
}

// Same fragment, sourced from f32 memory with on-the-fly f16 convert and
// zero padding for k >= kvalid (k measured relative to `base` column).
__device__ inline v16h frag_f32(const float* base, int ld, int kvalid) {
  int lane = threadIdx.x & 31;
  int r    = lane & 15;
  int kb   = (lane & 16) ? 8 : 0;
  v16h v;
#pragma unroll
  for (int i = 0; i < 8; ++i) {
    int k = kb + i;
    v[i] = (k < kvalid) ? (_Float16)base[(long)r * ld + k] : (_Float16)0.f;
  }
#pragma unroll
  for (int i = 0; i < 8; ++i) {
    int k = 16 + kb + i;
    v[8 + i] = (k < kvalid) ? (_Float16)base[(long)r * ld + k] : (_Float16)0.f;
  }
  return v;
}

// ---------------------------------------------------------------------------
// Kernel 1: gather word embeddings -> X[:, 0:300] (f16), zero pad [428:448).
// Columns [300:428) are written later by the char-LSTM kernels.
// ---------------------------------------------------------------------------
__global__ void k_gather_emb(const int* __restrict__ sent,
                             const float* __restrict__ emb,
                             _Float16* __restrict__ X) {
  int r = blockIdx.x;
  long tok = (long)sent[r];
  const float* e = emb + tok * E_;
  for (int c = threadIdx.x; c < KPAD; c += blockDim.x) {
    if (c < E_)         X[(long)r * KPAD + c] = (_Float16)e[c];
    else if (c >= Ein_) X[(long)r * KPAD + c] = (_Float16)0.f;
  }
}

// ---------------------------------------------------------------------------
// Kernel 2: char BiLSTM (one direction). 16 sequences per block, 4 waves.
// Each wave owns a 16-column j-tile and computes its i/f/g/o gate tiles so the
// cell update stays in registers; h lives in LDS (f16) for the next step's
// WMMA A-fragment. Final hidden -> X[:, 300:364) (bwd) or [364:428) (fwd).
// ---------------------------------------------------------------------------
__global__ __launch_bounds__(128) void k_char_lstm(
    const int* __restrict__ char_seq, const float* __restrict__ char_emb,
    const float* __restrict__ Wih, const float* __restrict__ Whh,
    const float* __restrict__ bih, const float* __restrict__ bhh,
    _Float16* __restrict__ X, int reverse) {
  __shared__ _Float16 ce[16 * 16 * 64];  // [seq][t][k]  32 KB
  __shared__ _Float16 hs[16 * 64];       // h state      2 KB

  int tid = threadIdx.x;
  int s0  = blockIdx.x * 16;

  for (int idx = tid; idx < 16 * 16 * 64; idx += 128) {
    int k = idx & 63, t = (idx >> 6) & 15, s = idx >> 10;
    int cid = char_seq[(s0 + s) * C_ + t];
    ce[idx] = (_Float16)char_emb[(long)cid * Ce_ + k];
  }
  for (int idx = tid; idx < 16 * 64; idx += 128) hs[idx] = (_Float16)0.f;
  __syncthreads();

  int lane = tid & 31;
  int jt   = tid >> 5;           // wave id == j-tile (0..3)
  int nl   = lane & 15;
  int mo   = (lane & 16) ? 8 : 0;

  float creg[8];
#pragma unroll
  for (int i = 0; i < 8; ++i) creg[i] = 0.f;

  float bias[4];
#pragma unroll
  for (int g = 0; g < 4; ++g) {
    int n = g * Hc_ + jt * 16 + nl;
    bias[g] = bih[n] + bhh[n];
  }

  for (int step = 0; step < 16; ++step) {
    int t = reverse ? (15 - step) : step;
    v8f acc[4];
#pragma unroll
    for (int g = 0; g < 4; ++g) acc[g] = zero8();

    // x_t contribution (K = Ce = 64 -> 2 chunks)
#pragma unroll
    for (int kc = 0; kc < 2; ++kc) {
      v16h a = frag_h(&ce[t * 64 + kc * 32], 16 * 64);
#pragma unroll
      for (int g = 0; g < 4; ++g) {
        int n0 = g * Hc_ + jt * 16;
        v16h b = frag_f32(Wih + (long)n0 * Ce_ + kc * 32, Ce_, 1 << 20);
        acc[g] = WMMA_F16(a, b, acc[g]);
      }
    }
    // h_{t-1} contribution (K = Hc = 64 -> 2 chunks)
#pragma unroll
    for (int kc = 0; kc < 2; ++kc) {
      v16h a = frag_h(&hs[kc * 32], 64);
#pragma unroll
      for (int g = 0; g < 4; ++g) {
        int n0 = g * Hc_ + jt * 16;
        v16h b = frag_f32(Whh + (long)n0 * Hc_ + kc * 32, Hc_, 1 << 20);
        acc[g] = WMMA_F16(a, b, acc[g]);
      }
    }
    __syncthreads();  // all reads of hs done before overwrite

#pragma unroll
    for (int i = 0; i < 8; ++i) {
      float ig = sigmoidf(acc[0][i] + bias[0]);
      float fg = sigmoidf(acc[1][i] + bias[1]);
      float gg = tanh_fast(acc[2][i] + bias[2]);
      float og = sigmoidf(acc[3][i] + bias[3]);
      creg[i] = fg * creg[i] + ig * gg;
      float h = og * tanh_fast(creg[i]);
      hs[(mo + i) * 64 + jt * 16 + nl] = (_Float16)h;
    }
    __syncthreads();
  }

  // write final hidden into the padded word-LSTM input
  int colbase = reverse ? E_ : (E_ + Hc_);  // hb at 300, hf at 364
#pragma unroll
  for (int i = 0; i < 8; ++i) {
    int m = mo + i;
    X[(long)(s0 + m) * KPAD + colbase + jt * 16 + nl] = hs[m * 64 + jt * 16 + nl];
  }
}

// ---------------------------------------------------------------------------
// Kernel 3: f32 -> f16 convert (used for Whh so the recurrent kernel can
// stream L2-resident f16 B-fragments directly from global).
// ---------------------------------------------------------------------------
__global__ void k_f32_to_f16(const float* __restrict__ src,
                             _Float16* __restrict__ dst, int n) {
  int i = blockIdx.x * blockDim.x + threadIdx.x;
  if (i < n) dst[i] = (_Float16)src[i];
}

// ---------------------------------------------------------------------------
// Kernel 4: XG = X (8192x448 f16) @ Wih^T (1024x428 f32->f16) + (bih+bhh).
// Bulk of the FLOPs; fully parallel WMMA GEMM. One 16x16 C-tile per wave.
// Output layout [t][b][n]: each timestep's 32x1024 slab is contiguous (128 KB)
// so the serial recurrent kernel reads L2-hot, prefetchable slabs.
// ---------------------------------------------------------------------------
__global__ __launch_bounds__(256) void k_word_xg(
    const _Float16* __restrict__ X, const float* __restrict__ Wih,
    const float* __restrict__ bih, const float* __restrict__ bhh,
    float* __restrict__ XG) {
  int wid  = threadIdx.x >> 5;
  int lane = threadIdx.x & 31;
  int mt   = wid & 1;
  int nt   = wid >> 1;
  int m0   = blockIdx.x * 32 + mt * 16;
  int n0   = blockIdx.y * 64 + nt * 16;

  v8f acc = zero8();
#pragma unroll 2
  for (int kc = 0; kc < KPAD / 32; ++kc) {
    v16h a = frag_h(X + (long)m0 * KPAD + kc * 32, KPAD);
    v16h b = frag_f32(Wih + (long)n0 * Ein_ + kc * 32, Ein_, Ein_ - kc * 32);
    acc = WMMA_F16(a, b, acc);
  }
  int n    = n0 + (lane & 15);
  float bb = bih[n] + bhh[n];
  int mo   = (lane & 16) ? 8 : 0;
#pragma unroll
  for (int i = 0; i < 8; ++i) {
    int r = m0 + mo + i;          // word row = b*T + t
    int b = r >> 8;               // / T_
    int t = r & (T_ - 1);         // % T_
    XG[((long)t * B_ + b) * GW_ + n] = acc[i] + bb;
  }
}

// ---------------------------------------------------------------------------
// Kernel 5: word LSTM recurrence (one direction). Single 1024-thread block
// (32 waves). Wave (mt, jt) owns rows mt*16..+16 and j-columns jt*16..+16 of
// all four gates -> cell state stays in registers, h (32x256 f16) in LDS.
// Whh B-fragments are loop-invariant -> compiler hoists them into VGPRs.
// ---------------------------------------------------------------------------
__global__ __launch_bounds__(1024) void k_word_rec(
    const float* __restrict__ XG, const _Float16* __restrict__ Whh16,
    _Float16* __restrict__ HCAT, int reverse, int dirOff) {
  __shared__ _Float16 hs[32 * 256];  // 16 KB

  int tid = threadIdx.x;
  for (int i = tid; i < 32 * 256; i += 1024) hs[i] = (_Float16)0.f;
  __syncthreads();

  int lane = tid & 31;
  int wid  = tid >> 5;   // 0..31
  int mt   = wid >> 4;   // 0..1
  int jt   = wid & 15;   // 0..15
  int nl   = lane & 15;
  int mo   = (lane & 16) ? 8 : 0;
  int m0   = mt * 16;

  float creg[8];
#pragma unroll
  for (int i = 0; i < 8; ++i) creg[i] = 0.f;

  for (int step = 0; step < T_; ++step) {
    int t = reverse ? (T_ - 1 - step) : step;
    const float* xgt = XG + (long)t * B_ * GW_;   // this step's 128 KB slab

    v8f acc[4];
#pragma unroll
    for (int g = 0; g < 4; ++g) {
      int n = g * Hw_ + jt * 16 + nl;
#pragma unroll
      for (int i = 0; i < 8; ++i)
        acc[g][i] = xgt[(long)(m0 + mo + i) * GW_ + n];
    }
    // prefetch next step's XG slab (gfx1250 global_prefetch_b8)
    if (step + 1 < T_) {
      int tn = reverse ? (t - 1) : (t + 1);
      __builtin_prefetch(&XG[(long)tn * B_ * GW_ + (long)(m0 + mo) * GW_ + jt * 16 + nl], 0, 1);
    }

#pragma unroll
    for (int kc = 0; kc < 8; ++kc) {  // K = Hw = 256
      v16h a = frag_h(&hs[m0 * 256 + kc * 32], 256);
#pragma unroll
      for (int g = 0; g < 4; ++g) {
        int n0 = g * Hw_ + jt * 16;
        v16h b = frag_h(Whh16 + (long)n0 * Hw_ + kc * 32, Hw_);
        acc[g] = WMMA_F16(a, b, acc[g]);
      }
    }
    __syncthreads();  // all reads of hs done

#pragma unroll
    for (int i = 0; i < 8; ++i) {
      float ig = sigmoidf(acc[0][i]);
      float fg = sigmoidf(acc[1][i]);
      float gg = tanh_fast(acc[2][i]);
      float og = sigmoidf(acc[3][i]);
      creg[i]  = fg * creg[i] + ig * gg;
      float h  = og * tanh_fast(creg[i]);
      int b = m0 + mo + i;
      int j = jt * 16 + nl;
      _Float16 h16 = (_Float16)h;
      hs[b * 256 + j] = h16;
      HCAT[((long)b * T_ + t) * 512 + dirOff + j] = h16;
    }
    __syncthreads();
  }
}

// ---------------------------------------------------------------------------
// Kernel 6: feats = HCAT (8192x512 f16) @ proj_W^T (32x512) + proj_b.
// ---------------------------------------------------------------------------
__global__ __launch_bounds__(64) void k_proj(
    const _Float16* __restrict__ HCAT, const float* __restrict__ projW,
    const float* __restrict__ projb, float* __restrict__ FEATS) {
  int wid  = threadIdx.x >> 5;
  int lane = threadIdx.x & 31;
  int m0   = blockIdx.x * 16;
  int n0   = wid * 16;

  v8f acc = zero8();
#pragma unroll 4
  for (int kc = 0; kc < 16; ++kc) {
    v16h a = frag_h(HCAT + (long)m0 * 512 + kc * 32, 512);
    v16h b = frag_f32(projW + (long)n0 * 512 + kc * 32, 512, 1 << 20);
    acc = WMMA_F16(a, b, acc);
  }
  int n    = n0 + (lane & 15);
  float bb = projb[n];
  int mo   = (lane & 16) ? 8 : 0;
#pragma unroll
  for (int i = 0; i < 8; ++i)
    FEATS[(long)(m0 + mo + i) * L_ + n] = acc[i] + bb;
}

// ---------------------------------------------------------------------------
// Kernel 7: Viterbi. One wave per batch element; lane == tag. fv in LDS,
// backpointers to workspace, sequential backtrack by lane 0.
// ---------------------------------------------------------------------------
__global__ __launch_bounds__(32) void k_viterbi(
    const float* __restrict__ FEATS, const float* __restrict__ trans,
    unsigned char* __restrict__ BP, float* __restrict__ out_score,
    float* __restrict__ out_path) {
  __shared__ float tr[L_ * L_];
  __shared__ float fv[L_];
  __shared__ float nfv[L_];

  int b = blockIdx.x;
  int i = threadIdx.x;
  for (int idx = i; idx < L_ * L_; idx += 32) tr[idx] = trans[idx];
  fv[i] = -10000.0f;
  __syncthreads();

  for (int t = 0; t < T_; ++t) {
    float m = -INFINITY;
    int bp = 0;
#pragma unroll
    for (int j = 0; j < L_; ++j) {
      float s = fv[j] + tr[i * L_ + j];
      if (s > m) { m = s; bp = j; }  // strict > keeps first max (jnp.argmax)
    }
    nfv[i] = m + FEATS[((long)b * T_ + t) * L_ + i];
    BP[(long)b * T_ * L_ + t * L_ + i] = (unsigned char)bp;
    __syncthreads();
    fv[i] = nfv[i];
    __syncthreads();
  }

  if (i == 0) {
    float best = fv[0];
    int last = 0;
    for (int j = 1; j < L_; ++j)
      if (fv[j] > best) { best = fv[j]; last = j; }
    out_score[b] = best;
    int tag = last;
    out_path[b * T_ + (T_ - 1)] = (float)tag;
    for (int t = T_ - 1; t >= 1; --t) {
      tag = BP[(long)b * T_ * L_ + t * L_ + tag];
      out_path[b * T_ + (t - 1)] = (float)tag;
    }
  }
}

// ---------------------------------------------------------------------------
extern "C" void kernel_launch(void* const* d_in, const int* in_sizes, int n_in,
                              void* d_out, int out_size, void* d_ws,
                              size_t ws_size, hipStream_t stream) {
  (void)in_sizes; (void)n_in; (void)out_size; (void)ws_size;

  const int*   sentence = (const int*)d_in[0];
  const int*   char_seq = (const int*)d_in[1];
  const float* emb      = (const float*)d_in[2];
  const float* char_emb = (const float*)d_in[3];
  const float* cWih_f   = (const float*)d_in[4];
  const float* cWhh_f   = (const float*)d_in[5];
  const float* cbih_f   = (const float*)d_in[6];
  const float* cbhh_f   = (const float*)d_in[7];
  const float* cWih_b   = (const float*)d_in[8];
  const float* cWhh_b   = (const float*)d_in[9];
  const float* cbih_b   = (const float*)d_in[10];
  const float* cbhh_b   = (const float*)d_in[11];
  const float* wWih_f   = (const float*)d_in[12];
  const float* wWhh_f   = (const float*)d_in[13];
  const float* wbih_f   = (const float*)d_in[14];
  const float* wbhh_f   = (const float*)d_in[15];
  const float* wWih_b   = (const float*)d_in[16];
  const float* wWhh_b   = (const float*)d_in[17];
  const float* wbih_b   = (const float*)d_in[18];
  const float* wbhh_b   = (const float*)d_in[19];
  const float* proj_W   = (const float*)d_in[20];
  const float* proj_b   = (const float*)d_in[21];
  const float* trans    = (const float*)d_in[22];

  char* ws = (char*)d_ws;
  size_t off = 0;
  auto alloc = [&](size_t bytes) -> void* {
    void* p = ws + off;
    off = (off + bytes + 255) & ~(size_t)255;
    return p;
  };
  _Float16*      X     = (_Float16*)alloc((size_t)BT * KPAD * 2);
  _Float16*      WHHF  = (_Float16*)alloc((size_t)GW_ * Hw_ * 2);
  _Float16*      WHHB  = (_Float16*)alloc((size_t)GW_ * Hw_ * 2);
  float*         XG    = (float*)alloc((size_t)BT * GW_ * 4);   // reused per dir
  _Float16*      HCAT  = (_Float16*)alloc((size_t)BT * 512 * 2);
  float*         FEATS = (float*)alloc((size_t)BT * L_ * 4);
  unsigned char* BP    = (unsigned char*)alloc((size_t)B_ * T_ * L_);

  // 1. word-embedding gather + pad
  k_gather_emb<<<BT, 128, 0, stream>>>(sentence, emb, X);

  // 2. char BiLSTM final hiddens -> X[:,300:428)
  k_char_lstm<<<BT / 16, 128, 0, stream>>>(char_seq, char_emb, cWih_b, cWhh_b,
                                           cbih_b, cbhh_b, X, 1);
  k_char_lstm<<<BT / 16, 128, 0, stream>>>(char_seq, char_emb, cWih_f, cWhh_f,
                                           cbih_f, cbhh_f, X, 0);

  // 3. pre-convert word Whh to f16 (L2-resident weight streaming)
  int nW = GW_ * Hw_;
  k_f32_to_f16<<<(nW + 255) / 256, 256, 0, stream>>>(wWhh_f, WHHF, nW);
  k_f32_to_f16<<<(nW + 255) / 256, 256, 0, stream>>>(wWhh_b, WHHB, nW);

  // 4/5. word LSTM, forward then backward (XG buffer reused)
  dim3 gxg(BT / 32, GW_ / 64);
  k_word_xg<<<gxg, 256, 0, stream>>>(X, wWih_f, wbih_f, wbhh_f, XG);
  k_word_rec<<<1, 1024, 0, stream>>>(XG, WHHF, HCAT, 0, 0);
  k_word_xg<<<gxg, 256, 0, stream>>>(X, wWih_b, wbih_b, wbhh_b, XG);
  k_word_rec<<<1, 1024, 0, stream>>>(XG, WHHB, HCAT, 1, Hw_);

  // 6. projection to label feats
  k_proj<<<BT / 16, 64, 0, stream>>>(HCAT, proj_W, proj_b, FEATS);

  // 7. Viterbi decode -> d_out = [score(32) | path(32*256) as float]
  float* out = (float*)d_out;
  k_viterbi<<<B_, 32, 0, stream>>>(FEATS, trans, BP, out, out + B_);
}
